// MoELayer_61692910240216
// MI455X (gfx1250) — matrix operand order repaired and checked by reading, not compiled, not run on hardware
//
#include <hip/hip_runtime.h>
#include <math.h>

// ---------------------------------------------------------------------------
// MoE expert MLP: out[t] = gelu(x[t] @ W1[e] + b1[e]) @ W2[e] + b2[e]
// T=16384 tokens, D=1024, F=4096, E=8 experts. bf16 WMMA path for gfx1250,
// with double-buffered LDS tiles fed by async global->LDS copies (ASYNCcnt).
// ---------------------------------------------------------------------------

#define E_EXPERTS 8
#define DM 1024
#define FF 4096
#define TTOK 16384
#define MT 128                 // M tile (tokens) per block
#define NTT 128                // N tile per block
#define KC 32                  // K chunk (bf16 WMMA K)
#define PADT (TTOK + E_EXPERTS * MT)   // 17408 max padded rows

typedef __bf16 bf16;
typedef __bf16 v16bf __attribute__((ext_vector_type(16)));
typedef __bf16 bf16x8 __attribute__((ext_vector_type(8)));
typedef __bf16 bf16x4 __attribute__((ext_vector_type(4)));
typedef float  v8f   __attribute__((ext_vector_type(8)));
typedef int    i32x4v __attribute__((vector_size(16)));   // builtin's pointee type

// ---- async global->LDS staging (CDNA5) ------------------------------------
#if defined(__has_builtin)
#  if __has_builtin(__builtin_amdgcn_global_load_async_to_lds_b128)
#    define HAS_ASYNC_LDS 1
#  endif
#endif
#ifndef HAS_ASYNC_LDS
#  define HAS_ASYNC_LDS 0
#endif

__device__ __forceinline__ void cp16_g2l(const bf16* g, bf16* l) {
#if HAS_ASYNC_LDS
    __builtin_amdgcn_global_load_async_to_lds_b128(
        (__attribute__((address_space(1))) i32x4v*)g,
        (__attribute__((address_space(3))) i32x4v*)l, 0, 0);
#else
    *(uint4*)l = *(const uint4*)g;          // synchronous fallback
#endif
}

__device__ __forceinline__ void wait_async_lds() {
#if HAS_ASYNC_LDS
#  if __has_builtin(__builtin_amdgcn_s_wait_asynccnt)
    __builtin_amdgcn_s_wait_asynccnt(0);
#  else
    asm volatile("s_wait_asynccnt 0" ::: "memory");
#  endif
#endif
}

// Fast tanh-GELU: 0.5*(1+tanh(u)) == sigmoid(2u)  ->  one v_exp_f32.
__device__ __forceinline__ float gelu_f(float x) {
    float u = -1.5957691216057308f * (x + 0.044715f * x * x * x);
    return x / (1.0f + __expf(u));
}

// ---------------------------- routing kernels ------------------------------

__global__ __launch_bounds__(256) void moe_init(int* wsI, int* tok) {
    int i = blockIdx.x * 256 + threadIdx.x;
    if (i < 16) wsI[i] = 0;          // counts[8] + cursor[8]
    if (i < PADT) tok[i] = -1;       // pad slots = invalid token
}

__global__ __launch_bounds__(256) void moe_count(const int* __restrict__ idx,
                                                 int* counts) {
    int t = blockIdx.x * 256 + threadIdx.x;
    if (t < TTOK) atomicAdd(&counts[idx[t] & 7], 1);
}

__global__ void moe_offsets(const int* __restrict__ counts, int* offs) {
    if (threadIdx.x == 0 && blockIdx.x == 0) {
        int off = 0;
        for (int e = 0; e < E_EXPERTS; ++e) {
            offs[e] = off;
            off += (counts[e] + (MT - 1)) & ~(MT - 1);   // align segments to 128
        }
        offs[E_EXPERTS] = off;
    }
}

__global__ __launch_bounds__(256) void moe_scatter(const int* __restrict__ idx,
                                                   int* cursor,
                                                   const int* __restrict__ offs,
                                                   int* tok) {
    int t = blockIdx.x * 256 + threadIdx.x;
    if (t < TTOK) {
        int e = idx[t] & 7;
        int p = atomicAdd(&cursor[e], 1);
        tok[offs[e] + p] = t;
    }
}

// ------------------------- fp32 -> bf16 conversion -------------------------

__global__ __launch_bounds__(256) void moe_cvt(const float* __restrict__ src,
                                               bf16* __restrict__ dst, int n8) {
    int i = blockIdx.x * 256 + threadIdx.x;      // 8 elements per thread
    if (i < n8) {
        const float4* s = (const float4*)src + (size_t)i * 2;
        float4 a = s[0], b = s[1];
        bf16x8 o;
        o[0] = (bf16)a.x; o[1] = (bf16)a.y; o[2] = (bf16)a.z; o[3] = (bf16)a.w;
        o[4] = (bf16)b.x; o[5] = (bf16)b.y; o[6] = (bf16)b.z; o[7] = (bf16)b.w;
        *(bf16x8*)(dst + (size_t)i * 8) = o;
    }
}

// Gather token rows into compacted, expert-sorted bf16 activation matrix.
__global__ __launch_bounds__(256) void moe_gather(const float* __restrict__ X,
                                                  const int* __restrict__ tok,
                                                  bf16* __restrict__ Xg) {
    int r = blockIdx.x;                     // padded row
    int t = tok[r];
    int c = threadIdx.x * 4;                // 256 threads x 4 = 1024 = D
    bf16x4 o;
    if (t < 0) {
        o[0] = (bf16)0.0f; o[1] = (bf16)0.0f; o[2] = (bf16)0.0f; o[3] = (bf16)0.0f;
    } else {
        float4 v = *(const float4*)(X + (size_t)t * DM + c);
        o[0] = (bf16)v.x; o[1] = (bf16)v.y; o[2] = (bf16)v.z; o[3] = (bf16)v.w;
    }
    *(bf16x4*)(Xg + (size_t)r * DM + c) = o;
}

// ------------------------------ WMMA GEMM ----------------------------------
// Block tile: 128(M) x 128(N), K chunked by 32. 256 threads = 8 waves; wave w
// owns M-subtile w (16 rows) and all 8 N-subtiles (8x v8f accumulators).
// LDS tiles are double-buffered; chunk k+1 is fetched with async global->LDS
// copies while chunk k runs through 8 WMMAs, then s_wait_asynccnt + barrier.
// A fragment layout (bf16 16x32, ISA 7.12.2): lane = m + 16*khalf;
//   halves [0..7]  = K = khalf*8 + 0..7,  halves [8..15] = K = 16 + khalf*8 + 0..7.
// B fragment layout (bf16 32x16): lane = n + 16*khalf; halves same K order as A.

template <int KDIM, int LDA, int LDB, bool GELU>
__global__ __launch_bounds__(256) void moe_gemm(
    const bf16* __restrict__ A,      // [rows][LDA], expert-sorted rows
    const bf16* __restrict__ B,      // + e*KDIM*LDB : [KDIM][LDB]
    const float* __restrict__ bias,  // [E][LDB]
    bf16* __restrict__ Hout,         // GELU phase: [rows][LDB] bf16
    float* __restrict__ Yout,        // final phase: scattered [T][LDB] f32
    const int* __restrict__ offs,    // [9] padded expert segment starts
    const int* __restrict__ tok)     // [PADT] token id per row (-1 = pad)
{
    const int m0 = blockIdx.x * MT;
    const int n0 = blockIdx.y * NTT;
    const int total = offs[E_EXPERTS];
    if (m0 >= total) return;                       // uniform early-out

    int e = 0;
    while (e < E_EXPERTS - 1 && m0 >= offs[e + 1]) ++e;

    __shared__ __align__(16) bf16 As[2][MT][KC + 8];    // stride 40 halves
    __shared__ __align__(16) bf16 Bs[2][KC][NTT + 8];   // stride 136 halves
    __shared__ int tokS[MT];

    const int tid  = threadIdx.x;
    const int lane = tid & 31;
    const int w    = tid >> 5;       // wave id == M-subtile

    if (!GELU && tid < MT) tokS[tid] = tok[m0 + tid];

    v8f acc[8];
#pragma unroll
    for (int i = 0; i < 8; ++i)
        acc[i] = (v8f){0.f, 0.f, 0.f, 0.f, 0.f, 0.f, 0.f, 0.f};

    const bf16* Ablk = A + (size_t)m0 * LDA;
    const bf16* Bblk = B + (size_t)e * KDIM * LDB + n0;

    // Per-thread staging slots: A: 2 threads/row, 32B each; B: 8 threads/row.
    const int arow = tid >> 1, aseg = tid & 1;
    const int brow = tid >> 3, bseg = tid & 7;

    // Prologue: stage chunk 0 into buffer 0.
    {
        const bf16* ga = Ablk + (size_t)arow * LDA + aseg * 16;
        cp16_g2l(ga,     &As[0][arow][aseg * 16]);
        cp16_g2l(ga + 8, &As[0][arow][aseg * 16 + 8]);
        const bf16* gb = Bblk + (size_t)brow * LDB + bseg * 16;
        cp16_g2l(gb,     &Bs[0][brow][bseg * 16]);
        cp16_g2l(gb + 8, &Bs[0][brow][bseg * 16 + 8]);
    }
    wait_async_lds();
    __syncthreads();

    int cur = 0;
    for (int k0 = 0; k0 < KDIM; k0 += KC) {
        const int nxt = cur ^ 1;
        if (k0 + KC < KDIM) {       // async-prefetch next chunk (overlaps WMMA)
            const bf16* ga = Ablk + (size_t)arow * LDA + (k0 + KC) + aseg * 16;
            cp16_g2l(ga,     &As[nxt][arow][aseg * 16]);
            cp16_g2l(ga + 8, &As[nxt][arow][aseg * 16 + 8]);
            const bf16* gb = Bblk + (size_t)(k0 + KC + brow) * LDB + bseg * 16;
            cp16_g2l(gb,     &Bs[nxt][brow][bseg * 16]);
            cp16_g2l(gb + 8, &Bs[nxt][brow][bseg * 16 + 8]);
        }

        // A fragment for this wave's 16 rows.
        const int mrow  = w * 16 + (lane & 15);
        const int khalf = lane >> 4;
        bf16x8 alo = *(const bf16x8*)(&As[cur][mrow][khalf * 8]);
        bf16x8 ahi = *(const bf16x8*)(&As[cur][mrow][16 + khalf * 8]);
        v16bf a;
#pragma unroll
        for (int i = 0; i < 8; ++i) { a[i] = alo[i]; a[8 + i] = ahi[i]; }

#pragma unroll
        for (int nt = 0; nt < 8; ++nt) {
            v16bf bm;
#pragma unroll
            for (int i = 0; i < 8; ++i) { bm[i] = Bs[cur][khalf * 8 + i][nt * 16 + (lane & 15)]; bm[8 + i] = Bs[cur][16 + khalf * 8 + i][nt * 16 + (lane & 15)]; }
            acc[nt] = __builtin_amdgcn_wmma_f32_16x16x32_bf16(
                false, a, false, bm, (short)0, acc[nt], false, false);
        }

        wait_async_lds();           // next chunk landed in LDS
        __syncthreads();            // everyone done reading 'cur'
        cur = nxt;
    }

    // Epilogue. C/D layout: VGPR r, lanes 0-15: M=r, N=lane; lanes 16-31: M=8+r.
    const int mb = w * 16 + (lane >> 4) * 8;
    const int nc = n0 + (lane & 15);
#pragma unroll
    for (int nt = 0; nt < 8; ++nt) {
        const int ncol = nc + nt * 16;
        const float bv = bias[e * LDB + ncol];
#pragma unroll
        for (int r = 0; r < 8; ++r) {
            int mloc = mb + r;
            float v = acc[nt][r] + bv;
            if (GELU) {
                Hout[(size_t)(m0 + mloc) * LDB + ncol] = (bf16)gelu_f(v);
            } else {
                int t = tokS[mloc];
                if (t >= 0) Yout[(size_t)t * LDB + ncol] = v;
            }
        }
    }
}

// ------------------------------- launcher ----------------------------------

extern "C" void kernel_launch(void* const* d_in, const int* in_sizes, int n_in,
                              void* d_out, int out_size, void* d_ws, size_t ws_size,
                              hipStream_t stream) {
    const float* X  = (const float*)d_in[0];   // [T][D]
    const int*   EI = (const int*)d_in[1];     // [T] expert ids
    const float* W1 = (const float*)d_in[2];   // [E][D][F]
    const float* B1 = (const float*)d_in[3];   // [E][F]
    const float* W2 = (const float*)d_in[4];   // [E][F][D]
    const float* B2 = (const float*)d_in[5];   // [E][D]
    float* OUT = (float*)d_out;

    char* ws = (char*)d_ws;
    int* wsI    = (int*)ws;
    int* counts = wsI;            // 8 ints
    int* cursor = wsI + 8;        // 8 ints
    int* offs   = wsI + 16;       // 9 ints
    int* tok    = (int*)(ws + 256);                         // PADT ints
    bf16* Xg  = (bf16*)(ws + (size_t)131072);               // PADT x D bf16
    bf16* W1b = (bf16*)(ws + (size_t)36 * 1024 * 1024);     // E x D x F bf16
    bf16* W2b = (bf16*)(ws + (size_t)101 * 1024 * 1024);    // E x F x D bf16
    bf16* Hg  = (bf16*)(ws + (size_t)168 * 1024 * 1024);    // PADT x F bf16

    (void)in_sizes; (void)n_in; (void)out_size; (void)ws_size;

    // 1) routing
    moe_init<<<PADT / 256, 256, 0, stream>>>(wsI, tok);
    moe_count<<<TTOK / 256, 256, 0, stream>>>(EI, counts);
    moe_offsets<<<1, 1, 0, stream>>>(counts, offs);
    moe_scatter<<<TTOK / 256, 256, 0, stream>>>(EI, cursor, offs, tok);

    // 2) precision staging (bf16 weights live in the 192MB L2 across tiles)
    const int W8 = (E_EXPERTS * DM * FF) / 8;   // 4,194,304 threads, 8 elems each
    moe_cvt<<<W8 / 256, 256, 0, stream>>>(W1, W1b, W8);
    moe_cvt<<<W8 / 256, 256, 0, stream>>>(W2, W2b, W8);
    moe_gather<<<PADT, 256, 0, stream>>>(X, tok, Xg);

    // 3) H = gelu(Xg @ W1[e] + b1[e])   M=padded rows, K=1024, N=4096
    moe_gemm<DM, DM, FF, true><<<dim3(PADT / MT, FF / NTT), 256, 0, stream>>>(
        Xg, W1b, B1, Hg, nullptr, offs, tok);

    // 4) out[t] = Hg @ W2[e] + b2[e]    K=4096, N=1024, scatter by token id
    moe_gemm<FF, FF, DM, false><<<dim3(PADT / MT, DM / NTT), 256, 0, stream>>>(
        Hg, W2b, B2, nullptr, OUT, offs, tok);
}
